// BalancedSA_24670292148967
// MI455X (gfx1250) — compile-verified, matrix operand order checked by reading, not run on hardware
//
#include <hip/hip_runtime.h>
#include <hip/hip_bf16.h>
#include <math.h>

typedef _Float16 half_t;
typedef __attribute__((ext_vector_type(16))) _Float16 v16h;
typedef __attribute__((ext_vector_type(8)))  _Float16 v8h;
typedef __attribute__((ext_vector_type(8)))  float    v8f;

// ---------------- problem constants ----------------
constexpr int B_      = 2;
constexpr int N_PATCH = 2048;
constexpr int N_TABLE = 16;
constexpr int N_CLS   = 1;
constexpr int N_      = N_PATCH + N_TABLE + N_CLS;   // 2065
constexpr int NPAD    = 2080;                        // 130 * 16
constexpr int E_      = 512;
constexpr int H_      = 8;
constexpr int D_      = 64;
constexpr float SCALE = 0.125f;                      // D^-0.5
constexpr float EPS_  = 1e-6f;

// d_out layout (fp32): out | A | A_raw
constexpr size_t OFF_OUT  = 0;
constexpr size_t SZ_OUT   = (size_t)B_ * N_ * E_;
constexpr size_t OFF_A    = SZ_OUT;
constexpr size_t SZ_A     = (size_t)B_ * H_ * N_ * N_;
constexpr size_t OFF_ARAW = OFF_A + SZ_A;

// workspace layout (f16 elements)
constexpr size_t WS_XH    = 0;                                        // (B*NPAD, E)
constexpr size_t WS_WQKVT = WS_XH    + (size_t)B_ * NPAD * E_;        // (3E, E) = w_qkv^T
constexpr size_t WS_WOUTT = WS_WQKVT + (size_t)3 * E_ * E_;           // (E, E)  = w_out^T
constexpr size_t WS_Q     = WS_WOUTT + (size_t)E_ * E_;               // (B*H, NPAD, D)
constexpr size_t WS_K     = WS_Q     + (size_t)B_ * H_ * NPAD * D_;
constexpr size_t WS_VT    = WS_K     + (size_t)B_ * H_ * NPAD * D_;   // (B*H, D, NPAD)
constexpr size_t WS_OH    = WS_VT    + (size_t)B_ * H_ * NPAD * D_;   // (B*NPAD, E)

// ---------------- WMMA fragment helpers ----------------
// A-fragment (16x32 f16, M x K), ISA 7.12.2 layout:
// lane<16 : row=lane,    K = {k0..k0+7, k0+16..k0+23}
// lane>=16: row=lane-16, K pattern shifted by +8

// branch-free fast loader: two aligned 16B vector loads, no guards
__device__ __forceinline__ v16h frag_f16(const half_t* p, int ld, int row0, int k0) {
  int lane = threadIdx.x & 31;
  const half_t* rp = p + (size_t)(row0 + (lane & 15)) * ld + k0 + ((lane >> 4) << 3);
  v8h lo = *(const v8h*)rp;
  v8h hi = *(const v8h*)(rp + 16);
  return __builtin_shufflevector(lo, hi, 0, 1, 2, 3, 4, 5, 6, 7,
                                 8, 9, 10, 11, 12, 13, 14, 15);
}

// k-guarded tail loader (rows unguarded: overreads stay inside the allocation)
__device__ __forceinline__ v16h frag_f16_kg(const half_t* p, int ld, int row0,
                                            int k0, int kmax) {
  int lane = threadIdx.x & 31;
  const half_t* rp = p + (size_t)(row0 + (lane & 15)) * ld;
  int kb = k0 + ((lane >> 4) << 3);
  v16h f;
#pragma unroll
  for (int e = 0; e < 8; ++e) {
    int k1 = kb + e;
    int k2 = kb + 16 + e;
    f[e]     = (k1 < kmax) ? rp[k1] : (half_t)0.f;
    f[e + 8] = (k2 < kmax) ? rp[k2] : (half_t)0.f;
  }
  return f;
}

// fp32 source (ld may be odd -> scalar loads) with in-register f32->f16 convert
__device__ __forceinline__ v16h frag_f32(const float* p, int ld, int row0, int k0) {
  int lane = threadIdx.x & 31;
  const float* rp = p + (size_t)(row0 + (lane & 15)) * ld;
  int kb = k0 + ((lane >> 4) << 3);
  __builtin_prefetch(rp + kb + 32, 0, 1);        // next K-step (global_prefetch_b8)
  v16h f;
#pragma unroll
  for (int e = 0; e < 8; ++e) {
    f[e]     = (half_t)rp[kb + e];
    f[e + 8] = (half_t)rp[kb + 16 + e];
  }
  return f;
}

__device__ __forceinline__ v16h frag_f32_kg(const float* p, int ld, int row0,
                                            int k0, int kmax) {
  int lane = threadIdx.x & 31;
  const float* rp = p + (size_t)(row0 + (lane & 15)) * ld;
  int kb = k0 + ((lane >> 4) << 3);
  v16h f;
#pragma unroll
  for (int e = 0; e < 8; ++e) {
    int k1 = kb + e;
    int k2 = kb + 16 + e;
    f[e]     = (k1 < kmax) ? (half_t)rp[k1] : (half_t)0.f;
    f[e + 8] = (k2 < kmax) ? (half_t)rp[k2] : (half_t)0.f;
  }
  return f;
}

__device__ __forceinline__ v8f wmma_f16(v16h a, v16h b, v8f c) {
  return __builtin_amdgcn_wmma_f32_16x16x32_f16(false, a, false, b, (short)0, c,
                                                false, false);
}

// ---------------- conversion / staging kernels ----------------
__global__ void conv_x_pad(const float* __restrict__ x, half_t* __restrict__ xh) {
  size_t idx = (size_t)blockIdx.x * blockDim.x + threadIdx.x;
  size_t total = (size_t)B_ * NPAD * E_;
  if (idx >= total) return;
  int e = idx % E_;
  int n = (idx / E_) % NPAD;
  int b = idx / ((size_t)NPAD * E_);
  xh[idx] = (n < N_) ? (half_t)x[((size_t)b * N_ + n) * E_ + e] : (half_t)0.f;
}

// dst[j][i] = (half) src[i][j]   (src: rows x cols row-major)
__global__ void conv_transpose(const float* __restrict__ src, half_t* __restrict__ dst,
                               int rows, int cols) {
  size_t idx = (size_t)blockIdx.x * blockDim.x + threadIdx.x;
  if (idx >= (size_t)rows * cols) return;
  int i = idx % rows;
  int j = idx / rows;
  dst[(size_t)j * rows + i] = (half_t)src[(size_t)i * cols + j];
}

// ---------------- GEMM 1: QKV projection + head split ----------------
// wave computes 4 adjacent N-tiles (64 cols); A fragment loaded once per K-step
__global__ void qkv_gemm(const half_t* __restrict__ xh, const half_t* __restrict__ wT,
                         const float* __restrict__ bias, half_t* __restrict__ qb,
                         half_t* __restrict__ kb_, half_t* __restrict__ vT) {
  int wave  = threadIdx.x >> 5;
  int lane  = threadIdx.x & 31;
  int nt0   = (blockIdx.x * 4 + wave) * 4;    // first of 4 col tiles (96 total)
  int tileM = blockIdx.y * 16;
  v8f acc0 = {}, acc1 = {}, acc2 = {}, acc3 = {};
  for (int kk = 0; kk < E_; kk += 32) {
    v16h a  = frag_f16(xh, E_, tileM, kk);
    v16h b0 = frag_f16(wT, E_, (nt0 + 0) * 16, kk);
    v16h b1 = frag_f16(wT, E_, (nt0 + 1) * 16, kk);
    v16h b2 = frag_f16(wT, E_, (nt0 + 2) * 16, kk);
    v16h b3 = frag_f16(wT, E_, (nt0 + 3) * 16, kk);
    acc0 = wmma_f16(a, b0, acc0);
    acc1 = wmma_f16(a, b1, acc1);
    acc2 = wmma_f16(a, b2, acc2);
    acc3 = wmma_f16(a, b3, acc3);
  }
  int b_i   = tileM / NPAD;                   // tile never straddles a batch
  int nbase = tileM - b_i * NPAD + ((lane >> 4) << 3);
#pragma unroll
  for (int t = 0; t < 4; ++t) {
    const v8f& acc = (t == 0) ? acc0 : (t == 1) ? acc1 : (t == 2) ? acc2 : acc3;
    int col = (nt0 + t) * 16 + (lane & 15);
    float bcol = bias[col];
    int part = col >> 9;                      // 0:q 1:k 2:v
    int jj   = col & 511;
    int h    = jj >> 6;
    int d    = jj & 63;
    size_t bh = (size_t)b_i * H_ + h;
#pragma unroll
    for (int j = 0; j < 8; ++j) {
      int n = nbase + j;
      if (n >= N_) continue;
      float v = acc[j] + bcol;
      if (part == 0)      qb [(bh * NPAD + n) * D_ + d] = (half_t)v;
      else if (part == 1) kb_[(bh * NPAD + n) * D_ + d] = (half_t)v;
      else                vT [(bh * D_ + d) * NPAD + n] = (half_t)v;
    }
  }
}

// ---------------- GEMM 2: A_raw = SCALE * q k^T per (b,h) ----------------
// fully branch-free: staging buffers are NPAD-padded, store guard discards edges
__global__ void scores_gemm(const half_t* __restrict__ qb, const half_t* __restrict__ kb_,
                            float* __restrict__ araw_base) {
  int wave  = threadIdx.x >> 5;
  int lane  = threadIdx.x & 31;
  int ct0   = (blockIdx.x * 4 + wave) * 4;   // first of 4 col tiles (130 total)
  if (ct0 >= NPAD / 16) return;              // wave-uniform
  int tileR = blockIdx.y * 16;
  int bh    = blockIdx.z;
  const half_t* qp = qb  + (size_t)bh * NPAD * D_;
  const half_t* kp = kb_ + (size_t)bh * NPAD * D_;
  v8f acc[4] = {{}, {}, {}, {}};
#pragma unroll
  for (int kk = 0; kk < D_; kk += 32) {
    v16h a = frag_f16(qp, D_, tileR, kk);
#pragma unroll
    for (int t = 0; t < 4; ++t) {
      v16h b = frag_f16(kp, D_, (ct0 + t) * 16, kk);
      acc[t] = wmma_f16(a, b, acc[t]);
    }
  }
  float* outp = araw_base + (size_t)bh * N_ * N_;
  int rbase = tileR + ((lane >> 4) << 3);
#pragma unroll
  for (int t = 0; t < 4; ++t) {
    int col = (ct0 + t) * 16 + (lane & 15);
    if (col >= N_) continue;
#pragma unroll
    for (int j = 0; j < 8; ++j) {
      int row = rbase + j;
      if (row < N_) outp[(size_t)row * N_ + col] = acc[t][j] * SCALE;
    }
  }
}

// ---------------- balanced softmax (one block per row, LDS-cached exp) -------
__device__ __forceinline__ float block_reduce(float v, float* sbuf) {
  int t = threadIdx.x;
  sbuf[t] = v;
  __syncthreads();
  for (int s = 128; s > 0; s >>= 1) {
    if (t < s) sbuf[t] += sbuf[t + s];
    __syncthreads();
  }
  float r = sbuf[0];
  __syncthreads();
  return r;
}

__global__ void balanced_softmax(const float* __restrict__ araw_base,
                                 float* __restrict__ a_base) {
  __shared__ float sbuf[256];
  __shared__ float ecache[N_PATCH + N_TABLE];
  int rid = blockIdx.x;
  int bh  = rid / N_;
  int n   = rid % N_;
  const float* rowp = araw_base + (size_t)bh * N_ * N_ + (size_t)n * N_;
  float*       outp = a_base    + (size_t)bh * N_ * N_ + (size_t)n * N_;
  int t = threadIdx.x;

  // single global read pass: raw-sum + exp-sum, exp cached in LDS
  float sp = 0.f, zp = 0.f;
  for (int i = t; i < N_PATCH; i += 256) {
    float a = rowp[i];
    float e = expf(a);
    sp += a;
    zp += e;
    ecache[i] = e;
  }
  float st = 0.f, zt = 0.f;
  if (t < N_TABLE) {
    float a = rowp[N_PATCH + t];
    float e = expf(a);
    st = a;
    zt = e;
    ecache[N_PATCH + t] = e;
  }
  float cv = rowp[N_PATCH + N_TABLE];

  float mp   = block_reduce(sp, sbuf) / (float)N_PATCH;
  float sump = block_reduce(zp, sbuf);
  float mt   = block_reduce(st, sbuf) / (float)N_TABLE;
  float sumt = block_reduce(zt, sbuf);

  float ep = expf(mp), et = expf(mt), ec = expf(cv);
  float s  = ep + et + ec;
  float ip = (ep / s) / (sump + EPS_);
  float it = (et / s) / (sumt + EPS_);

  for (int i = t; i < N_PATCH; i += 256) outp[i] = ecache[i] * ip;
  if (t < N_TABLE) outp[N_PATCH + t] = ecache[N_PATCH + t] * it;
  if (t == 0)      outp[N_PATCH + N_TABLE] = ec / (ec + EPS_) * (ec / s);
}

// ---------------- GEMM 3: out_h = A @ V per (b,h) ----------------
// one wave computes ALL 4 d-tiles -> each A fragment (HBM-streamed) read once.
// Main K loop (kk<2048) branch-free; single k-guarded tail step covers 2048..2079.
__global__ void av_gemm(const float* __restrict__ a_base, const half_t* __restrict__ vT,
                        half_t* __restrict__ oh) {
  int wave = threadIdx.x >> 5;
  int lane = threadIdx.x & 31;
  int rt   = blockIdx.x * 4 + wave;           // row tile (130 total)
  if (rt >= NPAD / 16) return;                // wave-uniform
  int tileR = rt * 16;
  int bh    = blockIdx.y;
  const float*  ap = a_base + (size_t)bh * N_ * N_;
  const half_t* vp = vT     + (size_t)bh * D_ * NPAD;
  v8f acc[4] = {{}, {}, {}, {}};
  for (int kk = 0; kk < N_PATCH; kk += 32) {  // 64 branch-free iterations
    v16h a = frag_f32(ap, N_, tileR, kk);
#pragma unroll
    for (int t = 0; t < 4; ++t) {
      v16h b = frag_f16(vp, NPAD, t * 16, kk);
      acc[t] = wmma_f16(a, b, acc[t]);
    }
  }
  {                                           // guarded tail: k in [2048, 2080)
    v16h a = frag_f32_kg(ap, N_, tileR, N_PATCH, N_);
#pragma unroll
    for (int t = 0; t < 4; ++t) {
      v16h b = frag_f16_kg(vp, NPAD, t * 16, N_PATCH, N_);
      acc[t] = wmma_f16(a, b, acc[t]);
    }
  }
  int b_i = bh / H_;
  int h   = bh % H_;
  int rbase = tileR + ((lane >> 4) << 3);
#pragma unroll
  for (int t = 0; t < 4; ++t) {
    int d = t * 16 + (lane & 15);
#pragma unroll
    for (int j = 0; j < 8; ++j) {
      int n = rbase + j;
      if (n < N_)
        oh[((size_t)b_i * NPAD + n) * E_ + h * D_ + d] = (half_t)acc[t][j];
    }
  }
}

// ---------------- GEMM 4: out = oh @ w_out + b_out ----------------
__global__ void out_gemm(const half_t* __restrict__ oh, const half_t* __restrict__ wT,
                         const float* __restrict__ bias, float* __restrict__ out) {
  int wave  = threadIdx.x >> 5;
  int lane  = threadIdx.x & 31;
  int nt0   = (blockIdx.x * 4 + wave) * 4;    // first of 4 col tiles (32 total)
  int tileM = blockIdx.y * 16;
  v8f acc0 = {}, acc1 = {}, acc2 = {}, acc3 = {};
  for (int kk = 0; kk < E_; kk += 32) {
    v16h a  = frag_f16(oh, E_, tileM, kk);
    v16h b0 = frag_f16(wT, E_, (nt0 + 0) * 16, kk);
    v16h b1 = frag_f16(wT, E_, (nt0 + 1) * 16, kk);
    v16h b2 = frag_f16(wT, E_, (nt0 + 2) * 16, kk);
    v16h b3 = frag_f16(wT, E_, (nt0 + 3) * 16, kk);
    acc0 = wmma_f16(a, b0, acc0);
    acc1 = wmma_f16(a, b1, acc1);
    acc2 = wmma_f16(a, b2, acc2);
    acc3 = wmma_f16(a, b3, acc3);
  }
  int b_i   = tileM / NPAD;                   // tile never straddles a batch
  int nbase = tileM - b_i * NPAD + ((lane >> 4) << 3);
#pragma unroll
  for (int t = 0; t < 4; ++t) {
    const v8f& acc = (t == 0) ? acc0 : (t == 1) ? acc1 : (t == 2) ? acc2 : acc3;
    int col = (nt0 + t) * 16 + (lane & 15);
    float bcol = bias[col];
#pragma unroll
    for (int j = 0; j < 8; ++j) {
      int n = nbase + j;
      if (n < N_)
        out[((size_t)b_i * N_ + n) * E_ + col] = acc[j] + bcol;
    }
  }
}

// ---------------- host-side launch ----------------
extern "C" void kernel_launch(void* const* d_in, const int* in_sizes, int n_in,
                              void* d_out, int out_size, void* d_ws, size_t ws_size,
                              hipStream_t stream) {
  const float* x     = (const float*)d_in[0];
  const float* w_qkv = (const float*)d_in[1];
  const float* b_qkv = (const float*)d_in[2];
  const float* w_out = (const float*)d_in[3];
  const float* b_out = (const float*)d_in[4];
  float* out = (float*)d_out;

  half_t* ws    = (half_t*)d_ws;
  half_t* xh    = ws + WS_XH;
  half_t* wqkvT = ws + WS_WQKVT;
  half_t* woutT = ws + WS_WOUTT;
  half_t* qb    = ws + WS_Q;
  half_t* kb    = ws + WS_K;
  half_t* vT    = ws + WS_VT;
  half_t* oh    = ws + WS_OH;

  float* out_p  = out + OFF_OUT;
  float* a_p    = out + OFF_A;
  float* araw_p = out + OFF_ARAW;

  {
    size_t tot = (size_t)B_ * NPAD * E_;
    conv_x_pad<<<dim3((unsigned)((tot + 255) / 256)), dim3(256), 0, stream>>>(x, xh);
    size_t tw = (size_t)E_ * 3 * E_;
    conv_transpose<<<dim3((unsigned)((tw + 255) / 256)), dim3(256), 0, stream>>>(w_qkv, wqkvT, E_, 3 * E_);
    size_t to = (size_t)E_ * E_;
    conv_transpose<<<dim3((unsigned)((to + 255) / 256)), dim3(256), 0, stream>>>(w_out, woutT, E_, E_);
  }

  // QKV: (B*NPAD,512) x (512,1536); 96 col tiles, 16 per block
  qkv_gemm<<<dim3(6, (B_ * NPAD) / 16), dim3(128), 0, stream>>>(
      xh, wqkvT, b_qkv, qb, kb, vT);

  // scores: per (b,h) 2080x2080x64; 130 col tiles, 16 per block
  scores_gemm<<<dim3((NPAD / 16 + 15) / 16, NPAD / 16, B_ * H_), dim3(128), 0, stream>>>(
      qb, kb, araw_p);

  balanced_softmax<<<dim3(B_ * H_ * N_), dim3(256), 0, stream>>>(araw_p, a_p);

  // AV: per (b,h); 130 row tiles, 4 per block, each wave does all 4 d-tiles
  av_gemm<<<dim3((NPAD / 16 + 3) / 4, B_ * H_), dim3(128), 0, stream>>>(a_p, vT, oh);

  // out-proj: (B*NPAD,512) x (512,512); 32 col tiles, 16 per block
  out_gemm<<<dim3(2, (B_ * NPAD) / 16), dim3(128), 0, stream>>>(
      oh, woutT, b_out, out_p);
}
